// MatchBatchHGAT_24773371363433
// MI455X (gfx1250) — matrix operand order repaired and checked by reading, not compiled, not run on hardware
//
#include <hip/hip_runtime.h>
#include <math.h>

// ---------------- problem constants ----------------
#define BB 32
#define NN 512
#define TT 3
#define HH 8
#define F_IN 768
#define F_HID 64
#define F_OUT 64
#define NEG_SLOPE 0.2f

typedef _Float16 half8  __attribute__((ext_vector_type(8)));
typedef _Float16 v16h   __attribute__((ext_vector_type(16)));
typedef float    v8f    __attribute__((ext_vector_type(8)));

// ============================================================
// 1) emb (f32) -> f16 A matrix [M=B*N, K=768] row-major
// ============================================================
__global__ void cvt_emb_kernel(const float* __restrict__ src, _Float16* __restrict__ dst, int n) {
  int i = blockIdx.x * blockDim.x + threadIdx.x;
  if (i < n) dst[i] = (_Float16)src[i];
}

// ============================================================
// 2) pack w1 (T,H,768,64) -> WMMA B layout [t][ct=32][kc=24][lane=32][16]
//    B'[t][k][c], c = h*64+o ; element j of lane: k = kc*32 + (lane<16 ? j : 16+j)
// ============================================================
__global__ void pack_w1_kernel(const float* __restrict__ w1, _Float16* __restrict__ bp) {
  int gid = blockIdx.x * blockDim.x + threadIdx.x;
  const int TOTAL = TT * 32 * 24 * 32 * 16;
  if (gid >= TOTAL) return;
  int j    = gid % 16;  int tmp = gid / 16;
  int lane = tmp % 32;  tmp /= 32;
  int kc   = tmp % 24;  tmp /= 24;
  int ct   = tmp % 32;  int t = tmp / 32;
  int k = kc * 32 + ((lane < 16) ? j : 16 + j);
  int c = ct * 16 + (lane & 15);
  int h = c >> 6, o = c & 63;
  float v = w1[(((size_t)(t * HH + h) * F_IN) + k) * 64 + o];
  bp[((((size_t)t * 32 + ct) * 24 + kc) * 32 + lane) * 16 + j] = (_Float16)v;
}

// ============================================================
// 3) pack w2 (T,1,512,64) -> [t][ct=4][kc=16][lane][16]
// ============================================================
__global__ void pack_w2_kernel(const float* __restrict__ w2, _Float16* __restrict__ bp) {
  int gid = blockIdx.x * blockDim.x + threadIdx.x;
  const int TOTAL = TT * 4 * 16 * 32 * 16;
  if (gid >= TOTAL) return;
  int j    = gid % 16;  int tmp = gid / 16;
  int lane = tmp % 32;  tmp /= 32;
  int kc   = tmp % 16;  tmp /= 16;
  int ct   = tmp % 4;   int t = tmp / 4;
  int k = kc * 32 + ((lane < 16) ? j : 16 + j);
  int c = ct * 16 + (lane & 15);
  float v = w2[((size_t)t * 512 + k) * 64 + c];
  bp[((((size_t)t * 4 + ct) * 16 + kc) * 32 + lane) * 16 + j] = (_Float16)v;
}

// ============================================================
// 4) pack hp1 (f32 [B*N,512]) -> [b][ct=32][kc=16][lane][16] f16 (B-layout, K=m)
// ============================================================
__global__ void pack_hp1_kernel(const float* __restrict__ hp1, _Float16* __restrict__ bp) {
  int gid = blockIdx.x * blockDim.x + threadIdx.x; // 32<<18 total
  int j    = gid & 15;
  int lane = (gid >> 4) & 31;
  int kc   = (gid >> 9) & 15;
  int ct   = (gid >> 13) & 31;
  int b    = gid >> 18;
  int m = kc * 32 + ((lane < 16) ? j : 16 + j);
  int c = ct * 16 + (lane & 15);
  float v = hp1[((size_t)(b * NN) + m) * 512 + c];
  bp[((((size_t)b * 32 + ct) * 16 + kc) * 32 + lane) * 16 + j] = (_Float16)v;
}

// ============================================================
// 5) typed GEMM: C[row,col] = sum_t vt[row,t] * (A @ B_t)[row,col]
//    One wave -> one 32x16 C tile (two 16x16 M-subtiles sharing the
//    three per-type B tiles).  B is double-buffered in registers so
//    next-chunk loads overlap the current WMMAs.  M multiple of 32,
//    K multiple of 32, Nc multiple of 16.
// ============================================================
__global__ void gemm_typed_kernel(const _Float16* __restrict__ A,
                                  const _Float16* __restrict__ Bp,
                                  const float* __restrict__ vt,
                                  float* __restrict__ C,
                                  int K, int Nc) {
  const int KC = K >> 5;
  const int CT = Nc >> 4;
  int wid  = (blockIdx.x * blockDim.x + threadIdx.x) >> 5;
  int lane = threadIdx.x & 31;
  int ct = wid % CT;
  int mt = wid / CT;          // 32-row tile index

  const _Float16* arow0 = A + (size_t)(mt * 32 + (lane & 15)) * K + ((lane < 16) ? 0 : 16);
  const _Float16* arow1 = arow0 + (size_t)16 * K;
  const size_t bstride_t = (size_t)CT * KC * 32 * 16;
  const _Float16* bbase = Bp + (size_t)ct * KC * 512 + lane * 16;

  v8f acc00 = {}; v8f acc01 = {}; v8f acc02 = {};
  v8f acc10 = {}; v8f acc11 = {}; v8f acc12 = {};

  // prime B double-buffer
  v16h b0 = *(const v16h*)(bbase);
  v16h b1 = *(const v16h*)(bbase + bstride_t);
  v16h b2 = *(const v16h*)(bbase + 2 * bstride_t);

  for (int kc = 0; kc < KC; ++kc) {
    // prefetch next-chunk B into alternate registers (clamped on last iter)
    int kn = (kc + 1 < KC) ? kc + 1 : kc;
    v16h nb0 = *(const v16h*)(bbase + (size_t)kn * 512);
    v16h nb1 = *(const v16h*)(bbase + bstride_t + (size_t)kn * 512);
    v16h nb2 = *(const v16h*)(bbase + 2 * bstride_t + (size_t)kn * 512);

    half8 lo0 = *(const half8*)(arow0 + kc * 32);
    half8 hi0 = *(const half8*)(arow0 + kc * 32 + 8);
    half8 lo1 = *(const half8*)(arow1 + kc * 32);
    half8 hi1 = *(const half8*)(arow1 + kc * 32 + 8);
    v16h a0 = __builtin_shufflevector(lo0, hi0, 0,1,2,3,4,5,6,7,8,9,10,11,12,13,14,15);
    v16h a1 = __builtin_shufflevector(lo1, hi1, 0,1,2,3,4,5,6,7,8,9,10,11,12,13,14,15);

    // stream-prefetch A two chunks ahead
    __builtin_prefetch(arow0 + (kc + 2) * 32, 0, 0);
    __builtin_prefetch(arow1 + (kc + 2) * 32, 0, 0);

    acc00 = __builtin_amdgcn_wmma_f32_16x16x32_f16(false, a0, false, b0, (short)0, acc00, false, false);
    acc10 = __builtin_amdgcn_wmma_f32_16x16x32_f16(false, a1, false, b0, (short)0, acc10, false, false);
    acc01 = __builtin_amdgcn_wmma_f32_16x16x32_f16(false, a0, false, b1, (short)0, acc01, false, false);
    acc11 = __builtin_amdgcn_wmma_f32_16x16x32_f16(false, a1, false, b1, (short)0, acc11, false, false);
    acc02 = __builtin_amdgcn_wmma_f32_16x16x32_f16(false, a0, false, b2, (short)0, acc02, false, false);
    acc12 = __builtin_amdgcn_wmma_f32_16x16x32_f16(false, a1, false, b2, (short)0, acc12, false, false);

    b0 = nb0; b1 = nb1; b2 = nb2;
  }

  int col = ct * 16 + (lane & 15);
  int rb0 = mt * 32 + ((lane < 16) ? 0 : 8);
#pragma unroll
  for (int v = 0; v < 8; ++v) {
    int row = rb0 + v;
    float w0 = vt[row * 3 + 0], w1 = vt[row * 3 + 1], w2 = vt[row * 3 + 2];
    C[(size_t)row * Nc + col] = w0 * acc00[v] + w1 * acc01[v] + w2 * acc02[v];
    row += 16;
    w0 = vt[row * 3 + 0]; w1 = vt[row * 3 + 1]; w2 = vt[row * 3 + 2];
    C[(size_t)row * Nc + col] = w0 * acc10[v] + w1 * acc11[v] + w2 * acc12[v];
  }
}

// ============================================================
// 6) src/dst attention scores: src[bh,n] = hp[b,n,h,:] . a_src[h,:]
// ============================================================
__global__ void srcdst_kernel(const float* __restrict__ hp,
                              const float* __restrict__ a_src,
                              const float* __restrict__ a_dst,
                              float* __restrict__ src, float* __restrict__ dst, int H) {
  int idx = blockIdx.x * blockDim.x + threadIdx.x; // B*H*N
  int n = idx % NN;
  int h = (idx / NN) % H;
  int b = idx / (NN * H);
  const float* row = hp + ((size_t)(b * NN + n)) * (H * 64) + h * 64;
  float s = 0.f, d = 0.f;
#pragma unroll 8
  for (int o = 0; o < 64; ++o) {
    float v = row[o];
    s += v * a_src[h * 64 + o];
    d += v * a_dst[h * 64 + o];
  }
  src[(size_t)(b * H + h) * NN + n] = s;
  dst[(size_t)(b * H + h) * NN + n] = d;
}

// ============================================================
// 7) per-row masked-softmax stats: rowmax, 1/rowsum (one wave per row)
// ============================================================
__global__ void rowstat_kernel(const float* __restrict__ src, const float* __restrict__ dst,
                               const int* __restrict__ adj,
                               float* __restrict__ rmax, float* __restrict__ isum, int H) {
  int wid  = (blockIdx.x * blockDim.x + threadIdx.x) >> 5;
  int lane = threadIdx.x & 31;
  int n  = wid % NN;
  int bh = wid / NN;
  int b  = bh / H;
  float s = src[(size_t)bh * NN + n];
  const float* drow = dst + (size_t)bh * NN;
  const int* arow = adj + ((size_t)b * NN + n) * NN;
  float mx = -3.0e38f;
  for (int m = lane; m < NN; m += 32) {
    if (arow[m] > 0) {
      float e = s + drow[m];
      e = (e >= 0.f) ? e : NEG_SLOPE * e;
      mx = fmaxf(mx, e);
    }
  }
  for (int off = 16; off; off >>= 1) mx = fmaxf(mx, __shfl_xor(mx, off, 32));
  float sm = 0.f;
  for (int m = lane; m < NN; m += 32) {
    if (arow[m] > 0) {
      float e = s + drow[m];
      e = (e >= 0.f) ? e : NEG_SLOPE * e;
      sm += __expf(e - mx);
    }
  }
  for (int off = 16; off; off >>= 1) sm += __shfl_xor(sm, off, 32);
  if (lane == 0) { rmax[(size_t)bh * NN + n] = mx; isum[(size_t)bh * NN + n] = 1.f / sm; }
}

// ============================================================
// 8) GAT-1 aggregation: out1 = attn @ hp1 (+b1), fused transpose + ELU,
//    written as f16 layer-2 A matrix x2[b*N+n, h*64+o].
//    One wave per (b, h, n-tile of 16, o-tile of 16); attention A-tile
//    built in registers each K-chunk of 32; B double-buffered.
// ============================================================
__global__ void gat1_agg_kernel(const _Float16* __restrict__ hp1p,
                                const int* __restrict__ adj,
                                const float* __restrict__ src1,
                                const float* __restrict__ rmax1,
                                const float* __restrict__ isum1,
                                const float* __restrict__ dst1,
                                const float* __restrict__ b1,
                                _Float16* __restrict__ x2) {
  int wid  = (blockIdx.x * blockDim.x + threadIdx.x) >> 5;
  int lane = threadIdx.x & 31;
  int ot = wid & 3;
  int nt = (wid >> 2) & 31;
  int h  = (wid >> 7) & 7;
  int b  = wid >> 10;
  int r    = lane & 15;
  int node = nt * 16 + r;
  int bh   = b * HH + h;

  float s  = src1[(size_t)bh * NN + node];
  float rm = rmax1[(size_t)bh * NN + node];
  float is = isum1[(size_t)bh * NN + node];
  const float* drow = dst1 + (size_t)bh * NN;
  const int* arow = adj + ((size_t)b * NN + node) * NN;
  const _Float16* bbase = hp1p + ((size_t)b * 32 + (h * 4 + ot)) * 16 * 512 + lane * 16;

  v8f acc = {};
  v16h bm = *(const v16h*)(bbase);       // prime double buffer
  for (int kc = 0; kc < 16; ++kc) {
    int kn = (kc + 1 < 16) ? kc + 1 : kc;
    v16h nbm = *(const v16h*)(bbase + (size_t)kn * 512);

    int mb = kc * 32 + ((lane < 16) ? 0 : 16);
    float4 d0 = *(const float4*)(drow + mb);
    float4 d1 = *(const float4*)(drow + mb + 4);
    float4 d2 = *(const float4*)(drow + mb + 8);
    float4 d3 = *(const float4*)(drow + mb + 12);
    int4 q0 = *(const int4*)(arow + mb);
    int4 q1 = *(const int4*)(arow + mb + 4);
    int4 q2 = *(const int4*)(arow + mb + 8);
    int4 q3 = *(const int4*)(arow + mb + 12);
    float dv[16] = { d0.x,d0.y,d0.z,d0.w, d1.x,d1.y,d1.z,d1.w,
                     d2.x,d2.y,d2.z,d2.w, d3.x,d3.y,d3.z,d3.w };
    int   av[16] = { q0.x,q0.y,q0.z,q0.w, q1.x,q1.y,q1.z,q1.w,
                     q2.x,q2.y,q2.z,q2.w, q3.x,q3.y,q3.z,q3.w };
    v16h amat;
#pragma unroll
    for (int j = 0; j < 16; ++j) {
      float e = s + dv[j];
      e = (e >= 0.f) ? e : NEG_SLOPE * e;
      float p = (av[j] > 0) ? __expf(e - rm) * is : 0.f;
      amat[j] = (_Float16)p;
    }
    acc = __builtin_amdgcn_wmma_f32_16x16x32_f16(false, amat, false, bm, (short)0, acc, false, false);
    bm = nbm;
  }
  int o = ot * 16 + (lane & 15);
  float bias = b1[o];
  int nb = nt * 16 + ((lane < 16) ? 0 : 8);
#pragma unroll
  for (int v = 0; v < 8; ++v) {
    float val = acc[v] + bias;
    val = (val > 0.f) ? val : (__expf(val) - 1.f); // ELU
    x2[((size_t)b * NN + (nb + v)) * 512 + h * 64 + o] = (_Float16)val;
  }
}

// ============================================================
// 9) head: aggregation-2 (rows 0,1 only) + ELU + hadamard + MLP + log_softmax
//    one block per graph, 64 threads
// ============================================================
__global__ void head_kernel(const float* __restrict__ hp2, const int* __restrict__ adj,
                            const float* __restrict__ src2, const float* __restrict__ dst2,
                            const float* __restrict__ rm2, const float* __restrict__ is2,
                            const float* __restrict__ b2,
                            const float* __restrict__ fc1_w, const float* __restrict__ fc1_b,
                            const float* __restrict__ fc2_w, const float* __restrict__ fc2_b,
                            const float* __restrict__ fc3_w, const float* __restrict__ fc3_b,
                            float* __restrict__ out) {
  __shared__ float attn_sh[NN];
  __shared__ float LR[2][64];
  __shared__ float vbuf[64];
  __shared__ float h1[192];
  __shared__ float h2[64];
  __shared__ float sc[2];
  int b = blockIdx.x;
  int t = threadIdx.x; // 0..63
  for (int n = 0; n < 2; ++n) {
    float s  = src2[(size_t)b * NN + n];
    float rm = rm2[(size_t)b * NN + n];
    float is = is2[(size_t)b * NN + n];
    for (int m = t; m < NN; m += 64) {
      float e = s + dst2[(size_t)b * NN + m];
      e = (e >= 0.f) ? e : NEG_SLOPE * e;
      attn_sh[m] = (adj[((size_t)b * NN + n) * NN + m] > 0) ? __expf(e - rm) * is : 0.f;
    }
    __syncthreads();
    float acc = 0.f;
    for (int m = 0; m < NN; ++m) acc += attn_sh[m] * hp2[((size_t)b * NN + m) * 64 + t];
    acc += b2[t];
    acc = (acc > 0.f) ? acc : (__expf(acc) - 1.f); // ELU
    LR[n][t] = acc;
    __syncthreads();
  }
  vbuf[t] = LR[0][t] * LR[1][t];
  __syncthreads();
  for (int j = t; j < 192; j += 64) {
    float a = fc1_b[j];
    for (int k = 0; k < 64; ++k) a += vbuf[k] * fc1_w[k * 192 + j];
    h1[j] = fmaxf(a, 0.f);
  }
  __syncthreads();
  {
    float a = fc2_b[t];
    for (int j = 0; j < 192; ++j) a += h1[j] * fc2_w[j * 64 + t];
    h2[t] = fmaxf(a, 0.f);
  }
  __syncthreads();
  if (t < 2) {
    float a = fc3_b[t];
    for (int o = 0; o < 64; ++o) a += h2[o] * fc3_w[o * 2 + t];
    sc[t] = a;
  }
  __syncthreads();
  if (t == 0) {
    float m = fmaxf(sc[0], sc[1]);
    float lse = m + logf(__expf(sc[0] - m) + __expf(sc[1] - m));
    out[b * 2 + 0] = sc[0] - lse;
    out[b * 2 + 1] = sc[1] - lse;
  }
}

// ============================================================
// host launcher
// ============================================================
extern "C" void kernel_launch(void* const* d_in, const int* in_sizes, int n_in,
                              void* d_out, int out_size, void* d_ws, size_t ws_size,
                              hipStream_t stream) {
  const float* emb    = (const float*)d_in[0];
  const int*   adj    = (const int*)  d_in[1];
  const float* vtypes = (const float*)d_in[2];
  const float* w1     = (const float*)d_in[3];
  const float* a_src1 = (const float*)d_in[4];
  const float* a_dst1 = (const float*)d_in[5];
  const float* b1     = (const float*)d_in[6];
  const float* w2     = (const float*)d_in[7];
  const float* a_src2 = (const float*)d_in[8];
  const float* a_dst2 = (const float*)d_in[9];
  const float* b2     = (const float*)d_in[10];
  const float* fc1_w  = (const float*)d_in[11];
  const float* fc1_b  = (const float*)d_in[12];
  const float* fc2_w  = (const float*)d_in[13];
  const float* fc2_b  = (const float*)d_in[14];
  const float* fc3_w  = (const float*)d_in[15];
  const float* fc3_b  = (const float*)d_in[16];
  float* out = (float*)d_out;

  // ---- workspace carve-up (all offsets 256-aligned) ----
  char* w = (char*)d_ws;
  size_t off = 0;
  auto alloc = [&](size_t bytes) { char* p = w + off; off += (bytes + 255) & ~(size_t)255; return p; };
  _Float16* embh  = (_Float16*)alloc((size_t)BB * NN * F_IN * 2);      // 25.2 MB
  _Float16* w1p   = (_Float16*)alloc((size_t)TT * F_IN * 512 * 2);     // 2.36 MB
  _Float16* w2p   = (_Float16*)alloc((size_t)TT * 512 * 64 * 2);       // 0.20 MB
  float*    hp1   = (float*)   alloc((size_t)BB * NN * 512 * 4);       // 33.5 MB
  float*    src1  = (float*)   alloc((size_t)BB * HH * NN * 4);
  float*    dst1  = (float*)   alloc((size_t)BB * HH * NN * 4);
  float*    rm1   = (float*)   alloc((size_t)BB * HH * NN * 4);
  float*    is1   = (float*)   alloc((size_t)BB * HH * NN * 4);
  _Float16* hp1p  = (_Float16*)alloc((size_t)BB * 512 * 512 * 2);      // 16.8 MB
  _Float16* x2    = (_Float16*)alloc((size_t)BB * NN * 512 * 2);       // 16.8 MB
  float*    hp2   = (float*)   alloc((size_t)BB * NN * 64 * 4);        // 4.2 MB
  float*    src2  = (float*)   alloc((size_t)BB * NN * 4);
  float*    dst2  = (float*)   alloc((size_t)BB * NN * 4);
  float*    rm2   = (float*)   alloc((size_t)BB * NN * 4);
  float*    is2   = (float*)   alloc((size_t)BB * NN * 4);
  (void)ws_size; (void)n_in; (void)in_sizes; (void)out_size;

  const int TPB = 256;

  // 1) emb -> f16
  {
    int n = BB * NN * F_IN;
    cvt_emb_kernel<<<n / TPB, TPB, 0, stream>>>(emb, embh, n);
  }
  // 2-3) pack weights
  pack_w1_kernel<<<(TT * 32 * 24 * 32 * 16) / TPB, TPB, 0, stream>>>(w1, w1p);
  pack_w2_kernel<<<(TT * 4 * 16 * 32 * 16) / TPB, TPB, 0, stream>>>(w2, w2p);
  // 4) layer-1 typed projection: hp1 = blend_t(embh @ w1[t])   (M=16384,K=768,Nc=512)
  {
    int waves = (BB * NN / 32) * (512 / 16); // 16384 (32-row tiles)
    gemm_typed_kernel<<<waves * 32 / TPB, TPB, 0, stream>>>(embh, w1p, vtypes, hp1, F_IN, 512);
  }
  // 5) attention scores + row stats (layer 1)
  srcdst_kernel<<<(BB * HH * NN) / TPB, TPB, 0, stream>>>(hp1, a_src1, a_dst1, src1, dst1, HH);
  rowstat_kernel<<<(BB * HH * NN) * 32 / TPB, TPB, 0, stream>>>(src1, dst1, adj, rm1, is1, HH);
  // 6) pack hp1 into WMMA B-layout
  pack_hp1_kernel<<<(BB << 18) / TPB, TPB, 0, stream>>>(hp1, hp1p);
  // 7) layer-1 aggregation fused with +b1, transpose, ELU -> x2 (f16)
  {
    int waves = BB * HH * (NN / 16) * 4; // 32768
    gat1_agg_kernel<<<waves * 32 / TPB, TPB, 0, stream>>>(hp1p, adj, src1, rm1, is1, dst1, b1, x2);
  }
  // 8) layer-2 typed projection: hp2 = blend_t(x2 @ w2[t])   (M=16384,K=512,Nc=64)
  {
    int waves = (BB * NN / 32) * (64 / 16); // 2048 (32-row tiles)
    gemm_typed_kernel<<<waves * 32 / TPB, TPB, 0, stream>>>(x2, w2p, vtypes, hp2, 512, 64);
  }
  // 9) attention scores + row stats (layer 2, H=1)
  srcdst_kernel<<<(BB * NN) / TPB, TPB, 0, stream>>>(hp2, a_src2, a_dst2, src2, dst2, 1);
  rowstat_kernel<<<(BB * NN) * 32 / TPB, TPB, 0, stream>>>(src2, dst2, adj, rm2, is2, 1);
  // 10) head: agg rows 0/1 + ELU + product + MLP + log_softmax
  head_kernel<<<BB, 64, 0, stream>>>(hp2, adj, src2, dst2, rm2, is2, b2,
                                     fc1_w, fc1_b, fc2_w, fc2_b, fc3_w, fc3_b, out);
}
